// LSTMModelBahdanau_72172630442702
// MI455X (gfx1250) — matrix operand order repaired and checked by reading, not compiled, not run on hardware
//
#include <hip/hip_runtime.h>
#include <hip/hip_bf16.h>

typedef __attribute__((ext_vector_type(16))) _Float16 v16h;
typedef __attribute__((ext_vector_type(8)))  _Float16 v8h;
typedef __attribute__((ext_vector_type(8)))  float    v8f;

#define Bsz 32
#define Tsz 256
#define IND 256
#define HID 512
#define ATTN 64
#define OUTD 7

// ---------------------------------------------------------------------------
// Wave-per-(16 x NT*16)-tile WMMA GEMM: C[M,N] = A[M,K] * W[N,K]^T (+bias)(+Cin)
// A, W row-major f16 with leading dim == K. Optionally batched via blockIdx.y.
// NT accumulators per wave: A fragment loaded once per k-step, reused NT times.
// ---------------------------------------------------------------------------
template <int NT>
__global__ __launch_bounds__(256) void wmma_gemm(
    const _Float16* __restrict__ A, long long sAb,
    const _Float16* __restrict__ W, long long sWb,
    const float* __restrict__ bias,
    const float* __restrict__ Cin, int cinRowStride,
    float* __restrict__ C, long long sCb,
    int M, int N, int K)
{
  const long long batch = blockIdx.y;
  A += batch * sAb;
  W += batch * sWb;
  C += batch * sCb;

  const int wave    = (blockIdx.x * blockDim.x + threadIdx.x) >> 5;
  const int ngroups = (N >> 4) / NT;               // N-tile groups of width NT*16
  const int m0 = (wave / ngroups) << 4;
  const int n0 = (wave % ngroups) * (NT * 16);
  if (m0 >= M) return;                 // wave-uniform: EXEC stays all-ones

  const int lane = threadIdx.x & 31;
  const int am  = m0 + (lane & 15);                // A row for this lane
  const int akb = (lane >> 4) << 3;                // A K sub-offset (0 or 8)
  const int bn  = n0 + (lane & 15);                // B col base for this lane
  const int bkb = (lane >> 4) << 3;                // B K sub-offset (0 or 8)

  v8f acc[NT];
#pragma unroll
  for (int j = 0; j < NT; ++j) acc[j] = (v8f){};

  for (int k0 = 0; k0 < K; k0 += 32) {
    // A fragment (shared across all NT output tiles)
    const _Float16* ap = A + (long long)am * K + (k0 + akb);
    v8h alo = *(const v8h*)(ap);       // K = base .. base+7
    v8h ahi = *(const v8h*)(ap + 16);  // K = base+16 .. base+23
    v16h af;
#pragma unroll
    for (int j = 0; j < 8; ++j) { af[j] = alo[j]; af[j + 8] = ahi[j]; }

#pragma unroll
    for (int jt = 0; jt < NT; ++jt) {
      const _Float16* bp = W + (long long)(bn + jt * 16) * K + (k0 + bkb);
      v8h blo = *(const v8h*)(bp);     // K = base .. base+7
      v8h bhi = *(const v8h*)(bp + 16);  // K = base+16 .. base+23
      v16h bf;
#pragma unroll
      for (int j = 0; j < 8; ++j) { bf[j] = blo[j]; bf[j + 8] = bhi[j]; }
      acc[jt] = __builtin_amdgcn_wmma_f32_16x16x32_f16(
          false, af, false, bf, (short)0, acc[jt], false, false);
    }
  }

  const int mb = m0 + ((lane >> 4) << 3);
#pragma unroll
  for (int jt = 0; jt < NT; ++jt) {
    const int n = n0 + jt * 16 + (lane & 15);
    const float bv = bias ? bias[n] : 0.0f;
#pragma unroll
    for (int r = 0; r < 8; ++r) {
      float val = acc[jt][r] + bv;
      if (Cin) val += Cin[(long long)(mb + r) * cinRowStride + n];
      C[(long long)(mb + r) * N + n] = val;
    }
  }
}

// ---------------------------------------------------------------------------
// Embedding gather -> f16
// ---------------------------------------------------------------------------
__global__ void embed_kernel(const int* __restrict__ ids,
                             const float* __restrict__ emb,
                             _Float16* __restrict__ X)
{
  int idx = blockIdx.x * blockDim.x + threadIdx.x;   // B*T*IND
  int k  = idx & (IND - 1);
  int bt = idx >> 8;
  X[idx] = (_Float16)emb[(long long)ids[bt] * IND + k];
}

// f32 -> f16 convert
__global__ void cvt_kernel(const float* __restrict__ s, _Float16* __restrict__ d, int n)
{
  int i = blockIdx.x * blockDim.x + threadIdx.x;
  if (i < n) d[i] = (_Float16)s[i];
}

// bias_g = bih + bhh ; bias_ed = concat(be, bd)
__global__ void bias_prep(const float* bih, const float* bhh,
                          const float* be, const float* bd,
                          float* bias_g, float* bias_ed)
{
  int i = blockIdx.x * blockDim.x + threadIdx.x;
  if (i < 4 * HID) bias_g[i] = bih[i] + bhh[i];
  if (i < ATTN)            bias_ed[i] = be[i];
  else if (i < 2 * ATTN)   bias_ed[i] = bd[i - ATTN];
}

__global__ void zero_state(_Float16* h16, float* c)
{
  int i = blockIdx.x * blockDim.x + threadIdx.x;
  if (i < Bsz * HID) { h16[i] = (_Float16)0.0f; c[i] = 0.0f; }
}

// ---------------------------------------------------------------------------
// LSTM per-step elementwise: gate nonlinearities, c/h update, h fan-out
// ---------------------------------------------------------------------------
__device__ inline float sigf(float x) { return 1.0f / (1.0f + __expf(-x)); }

__global__ void lstm_elem(const float* __restrict__ gates,
                          float* __restrict__ c,
                          _Float16* __restrict__ h16,
                          _Float16* __restrict__ lo16,
                          _Float16* __restrict__ loT16,
                          float* __restrict__ lo32, int t)
{
  int idx = blockIdx.x * blockDim.x + threadIdx.x;   // B*HID
  int b = idx >> 9, n = idx & (HID - 1);
  const float* g = gates + (long long)b * 4 * HID;
  float gi = g[n];
  float gf = g[HID + n];
  float gg = g[2 * HID + n];
  float go = g[3 * HID + n];
  float cn = sigf(gf) * c[idx] + sigf(gi) * tanhf(gg);
  float h  = sigf(go) * tanhf(cn);
  c[idx] = cn;
  h16[idx] = (_Float16)h;
  long long row = (long long)(b * Tsz + t);
  lo32[row * HID + n] = h;
  lo16[row * HID + n] = (_Float16)h;
  loT16[(long long)b * HID * Tsz + (long long)n * Tsz + t] = (_Float16)h;
}

// ---------------------------------------------------------------------------
// Attention: scores (tanh energy dot v) + softmax over te, output f16 weights
// One block per (b, td); thread = te.
// ---------------------------------------------------------------------------
__global__ __launch_bounds__(256) void attn_kernel(
    const float* __restrict__ EF,      // [B*T, 128]: cols 0..63 enc, 64..127 dec
    const float* __restrict__ v, const float* __restrict__ vb,
    _Float16* __restrict__ Wt)         // [B, Td, Te] f16
{
  int b = blockIdx.y, td = blockIdx.x, te = threadIdx.x;
  __shared__ float sdf[ATTN];
  __shared__ float sv[ATTN];
  __shared__ float red[Tsz];
  if (te < ATTN) {
    sdf[te] = EF[((long long)(b * Tsz + td)) * 128 + ATTN + te];
    sv[te]  = v[te];
  }
  __syncthreads();

  const float* ef = EF + ((long long)(b * Tsz + te)) * 128;
  float s = vb[0];
#pragma unroll 8
  for (int a = 0; a < ATTN; ++a) s += sv[a] * tanhf(sdf[a] + ef[a]);

  red[te] = s; __syncthreads();
  for (int off = 128; off; off >>= 1) {
    if (te < off) red[te] = fmaxf(red[te], red[te + off]);
    __syncthreads();
  }
  float mx = red[0]; __syncthreads();
  float e = __expf(s - mx);
  red[te] = e; __syncthreads();
  for (int off = 128; off; off >>= 1) {
    if (te < off) red[te] += red[te + off];
    __syncthreads();
  }
  float inv = 1.0f / red[0];
  Wt[((long long)b * Tsz + td) * Tsz + te] = (_Float16)(e * inv);
}

// ---------------------------------------------------------------------------
// Final FC: logits = [lstm_out, context] @ Wfc^T + bfc   (N=7, VALU)
// ---------------------------------------------------------------------------
__global__ void final_fc(const float* __restrict__ lo32,
                         const float* __restrict__ ctx,
                         const float* __restrict__ Wfc,
                         const float* __restrict__ bfc,
                         float* __restrict__ out)
{
  int idx = blockIdx.x * blockDim.x + threadIdx.x;
  if (idx >= Bsz * Tsz * OUTD) return;
  int o  = idx % OUTD;
  int bt = idx / OUTD;
  const float* hrow = lo32 + (long long)bt * HID;
  const float* crow = ctx  + (long long)bt * HID;
  const float* w = Wfc + (long long)o * 2 * HID;
  float acc = bfc[o];
  for (int h = 0; h < HID; ++h) acc += hrow[h] * w[h];
  for (int h = 0; h < HID; ++h) acc += crow[h] * w[HID + h];
  out[idx] = acc;
}

// ---------------------------------------------------------------------------
extern "C" void kernel_launch(void* const* d_in, const int* in_sizes, int n_in,
                              void* d_out, int out_size, void* d_ws, size_t ws_size,
                              hipStream_t stream)
{
  const int*   ids = (const int*)  d_in[0];
  const float* emb = (const float*)d_in[1];
  const float* Wih = (const float*)d_in[2];
  const float* Whh = (const float*)d_in[3];
  const float* bih = (const float*)d_in[4];
  const float* bhh = (const float*)d_in[5];
  const float* We  = (const float*)d_in[6];
  const float* be  = (const float*)d_in[7];
  const float* Wd  = (const float*)d_in[8];
  const float* bd  = (const float*)d_in[9];
  const float* v   = (const float*)d_in[10];
  const float* vb  = (const float*)d_in[11];
  const float* Wfc = (const float*)d_in[12];
  const float* bfc = (const float*)d_in[13];
  float* out = (float*)d_out;

  // workspace carve (256-byte aligned)
  char* w = (char*)d_ws;
  auto carve = [&](size_t bytes) -> void* {
    void* p = (void*)w;
    w += (bytes + 255) & ~(size_t)255;
    return p;
  };
  const long long BT = (long long)Bsz * Tsz;                 // 8192
  _Float16* Xf16   = (_Float16*)carve(BT * IND * 2);         // embedded input, f16
  _Float16* Wihf16 = (_Float16*)carve((size_t)4*HID*IND*2);
  _Float16* Whhf16 = (_Float16*)carve((size_t)4*HID*HID*2);
  _Float16* Wedf16 = (_Float16*)carve((size_t)2*ATTN*HID*2); // [We;Wd] rows
  float*    bias_g = (float*)   carve((size_t)4*HID*4);
  float*    bias_ed= (float*)   carve((size_t)2*ATTN*4);
  float*    G      = (float*)   carve(BT * 4*HID * 4);       // X@Wih^T + biases
  float*    gates  = (float*)   carve((size_t)Bsz*4*HID*4);
  _Float16* h16    = (_Float16*)carve((size_t)Bsz*HID*2);
  float*    c      = (float*)   carve((size_t)Bsz*HID*4);
  _Float16* lo16   = (_Float16*)carve(BT * HID * 2);         // lstm_out f16 [bt,h]
  _Float16* loT16  = (_Float16*)carve(BT * HID * 2);         // lstm_out f16 [b,h,t]
  float*    lo32   = (float*)   carve(BT * HID * 4);         // lstm_out f32
  float*    EF     = (float*)   carve(BT * 128 * 4);         // enc|dec features
  _Float16* Wt     = (_Float16*)carve(BT * Tsz * 2);         // softmax weights f16
  float*    ctx    = (float*)   carve(BT * HID * 4);         // context f32

  // --- prep: embedding gather, weight converts, bias fold, state zero ------
  embed_kernel<<<(BT * IND) / 256, 256, 0, stream>>>(ids, emb, Xf16);
  cvt_kernel<<<(4*HID*IND + 255)/256, 256, 0, stream>>>(Wih, Wihf16, 4*HID*IND);
  cvt_kernel<<<(4*HID*HID + 255)/256, 256, 0, stream>>>(Whh, Whhf16, 4*HID*HID);
  cvt_kernel<<<(ATTN*HID + 255)/256, 256, 0, stream>>>(We, Wedf16, ATTN*HID);
  cvt_kernel<<<(ATTN*HID + 255)/256, 256, 0, stream>>>(Wd, Wedf16 + ATTN*HID, ATTN*HID);
  bias_prep<<<(4*HID + 255)/256, 256, 0, stream>>>(bih, bhh, be, bd, bias_g, bias_ed);
  zero_state<<<(Bsz*HID + 255)/256, 256, 0, stream>>>(h16, c);

  // --- GEMM1: G = X @ Wih^T + (bih+bhh)   M=8192 N=2048 K=256, 16x64/wave --
  {
    int waves = (int)(BT / 16) * ((4*HID / 16) / 4);   // 16384
    wmma_gemm<4><<<dim3(waves / 8, 1), 256, 0, stream>>>(
        Xf16, 0, Wihf16, 0, bias_g, nullptr, 0, G, 0,
        (int)BT, 4*HID, IND);
  }

  // --- LSTM recurrence: 256 steps; 16x16/wave for max parallelism ---------
  for (int t = 0; t < Tsz; ++t) {
    int waves = (Bsz / 16) * (4*HID / 16);   // 256 waves
    wmma_gemm<1><<<dim3(waves / 8, 1), 256, 0, stream>>>(
        h16, 0, Whhf16, 0, nullptr,
        G + (long long)t * 4*HID, Tsz * 4*HID,   // Cin = G row-slice (stride T*4H)
        gates, 0, Bsz, 4*HID, HID);
    lstm_elem<<<(Bsz*HID + 255)/256, 256, 0, stream>>>(
        gates, c, h16, lo16, loT16, lo32, t);
  }

  // --- enc/dec features: EF = lstm_out @ [We;Wd]^T + [be;bd], 16x64/wave ---
  {
    int waves = (int)(BT / 16) * ((128 / 16) / 4);     // 1024
    wmma_gemm<4><<<dim3(waves / 8, 1), 256, 0, stream>>>(
        lo16, 0, Wedf16, 0, bias_ed, nullptr, 0, EF, 0,
        (int)BT, 2*ATTN, HID);
  }

  // --- attention scores + softmax -> f16 weights ---------------------------
  attn_kernel<<<dim3(Tsz, Bsz), Tsz, 0, stream>>>(EF, v, vb, Wt);

  // --- context: per-b  Wt[256,256] @ lstm[256,512] via lstm^T, 16x64/wave --
  {
    int waves = (Tsz / 16) * ((HID / 16) / 4);         // 128 per batch
    wmma_gemm<4><<<dim3(waves / 8, Bsz), 256, 0, stream>>>(
        Wt, (long long)Tsz * Tsz,
        loT16, (long long)HID * Tsz,
        nullptr, nullptr, 0,
        ctx, (long long)Tsz * HID,
        Tsz, HID, Tsz);
  }

  // --- final FC ------------------------------------------------------------
  final_fc<<<(Bsz*Tsz*OUTD + 255)/256, 256, 0, stream>>>(lo32, ctx, Wfc, bfc, out);
}
